// LSTMAttentive_8349416423883
// MI455X (gfx1250) — matrix operand order repaired and checked by reading, not compiled, not run on hardware
//
#include <hip/hip_runtime.h>
#include <hip/hip_bf16.h>
#include <math.h>

// Shapes (fixed by the reference): B=1024, T=64, H=1024, E=1024
static constexpr int Bc = 1024;
static constexpr int Tc = 64;
static constexpr int Hc = 1024;
static constexpr int Ec = 1024;

typedef __attribute__((ext_vector_type(16))) __bf16        bf16x16;
typedef __attribute__((ext_vector_type(8)))  __bf16        bf16x8;
typedef __attribute__((ext_vector_type(8)))  float         f32x8;
typedef __attribute__((ext_vector_type(4)))  unsigned int  u32x4;
typedef __attribute__((ext_vector_type(8)))  unsigned int  u32x8;

#define DEV __device__ __forceinline__

DEV float sigm(float x) { return 1.0f / (1.0f + expf(-x)); }

// ---------------------------------------------------------------------------
// WMMA operand loaders (contiguous k = 16*half + e in both A and B, wave32).
//
// A (16x32, MxK) from bf16 row-major (ld == K): lane = (half<<4)|r holds row r,
//   element e -> K = 16*half + e
DEV bf16x16 load_a_bf16(const __bf16* __restrict__ a, int ld, int lane) {
  const int r  = lane & 15;
  const int kh = (lane >> 4) << 4;
  const __bf16* p = a + (size_t)r * ld + kh;
  u32x4 lo = *(const u32x4*)(p);       // K = kh .. kh+7
  u32x4 hi = *(const u32x4*)(p + 8);   // K = kh+8 .. kh+15
  u32x8 r8;
  r8[0] = lo[0]; r8[1] = lo[1]; r8[2] = lo[2]; r8[3] = lo[3];
  r8[4] = hi[0]; r8[5] = hi[1]; r8[6] = hi[2]; r8[7] = hi[3];
  return __builtin_bit_cast(bf16x16, r8);
}

// A (16x32) from f32 row-major, converting in-register (for streamed features).
DEV bf16x16 load_a_f32(const float* __restrict__ base, int ld, int lane) {
  const int r  = lane & 15;
  const int kh = (lane >> 4) << 4;
  const float* p = base + (size_t)r * ld + kh;
  float4 x0 = *(const float4*)(p);
  float4 x1 = *(const float4*)(p + 4);
  float4 y0 = *(const float4*)(p + 8);
  float4 y1 = *(const float4*)(p + 12);
  bf16x16 t;
  t[0]  = (__bf16)x0.x; t[1]  = (__bf16)x0.y; t[2]  = (__bf16)x0.z; t[3]  = (__bf16)x0.w;
  t[4]  = (__bf16)x1.x; t[5]  = (__bf16)x1.y; t[6]  = (__bf16)x1.z; t[7]  = (__bf16)x1.w;
  t[8]  = (__bf16)y0.x; t[9]  = (__bf16)y0.y; t[10] = (__bf16)y0.z; t[11] = (__bf16)y0.w;
  t[12] = (__bf16)y1.x; t[13] = (__bf16)y1.y; t[14] = (__bf16)y1.z; t[15] = (__bf16)y1.w;
  return t;
}

// B (32x16, KxN) from weight W stored [N,K] row-major bf16 (C = A @ W^T):
// lane = (half<<4)|n holds col n, element e -> K = 16*half + e.
// 16 consecutive bf16 = one 32B vector load (two b128), straight into operand regs.
DEV bf16x16 load_b_bf16(const __bf16* __restrict__ w, int ld, int lane) {
  const int n  = lane & 15;
  const int kh = (lane >> 4) << 4;
  u32x8 r8 = *(const u32x8*)(w + (size_t)n * ld + kh);
  return __builtin_bit_cast(bf16x16, r8);
}

DEV f32x8 wmma_bf16(bf16x16 a, bf16x16 b, f32x8 c) {
  return __builtin_amdgcn_wmma_f32_16x16x32_bf16(false, a, false, b, (short)0, c,
                                                 false, false);
}

// Double-buffered K-loop: 16 rows x 64 cols per wave, A and W both bf16.
DEV void gemm_loop_bf16(const __bf16* __restrict__ A, const __bf16* __restrict__ W,
                        int K, int m0, int n0, int lane, f32x8 acc[4]) {
  const __bf16* arow = A + (size_t)m0 * K;
  const __bf16* w0 = W + (size_t)(n0)      * K;
  const __bf16* w1 = W + (size_t)(n0 + 16) * K;
  const __bf16* w2 = W + (size_t)(n0 + 32) * K;
  const __bf16* w3 = W + (size_t)(n0 + 48) * K;
  bf16x16 a  = load_a_bf16(arow, K, lane);
  bf16x16 b0 = load_b_bf16(w0, K, lane);
  bf16x16 b1 = load_b_bf16(w1, K, lane);
  bf16x16 b2 = load_b_bf16(w2, K, lane);
  bf16x16 b3 = load_b_bf16(w3, K, lane);
  for (int k = 32; k < K; k += 32) {
    bf16x16 an = load_a_bf16(arow + k, K, lane);
    bf16x16 c0 = load_b_bf16(w0 + k, K, lane);
    bf16x16 c1 = load_b_bf16(w1 + k, K, lane);
    bf16x16 c2 = load_b_bf16(w2 + k, K, lane);
    bf16x16 c3 = load_b_bf16(w3 + k, K, lane);
    acc[0] = wmma_bf16(a, b0, acc[0]);
    acc[1] = wmma_bf16(a, b1, acc[1]);
    acc[2] = wmma_bf16(a, b2, acc[2]);
    acc[3] = wmma_bf16(a, b3, acc[3]);
    a = an; b0 = c0; b1 = c1; b2 = c2; b3 = c3;
  }
  acc[0] = wmma_bf16(a, b0, acc[0]);
  acc[1] = wmma_bf16(a, b1, acc[1]);
  acc[2] = wmma_bf16(a, b2, acc[2]);
  acc[3] = wmma_bf16(a, b3, acc[3]);
}

// Same but A is streamed f32 (features), converted in-register.
DEV void gemm_loop_f32a(const float* __restrict__ A, const __bf16* __restrict__ W,
                        int K, int m0, int n0, int lane, f32x8 acc[4]) {
  const float* arow = A + (size_t)m0 * K;
  const __bf16* w0 = W + (size_t)(n0)      * K;
  const __bf16* w1 = W + (size_t)(n0 + 16) * K;
  const __bf16* w2 = W + (size_t)(n0 + 32) * K;
  const __bf16* w3 = W + (size_t)(n0 + 48) * K;
  bf16x16 a  = load_a_f32(arow, K, lane);
  bf16x16 b0 = load_b_bf16(w0, K, lane);
  bf16x16 b1 = load_b_bf16(w1, K, lane);
  bf16x16 b2 = load_b_bf16(w2, K, lane);
  bf16x16 b3 = load_b_bf16(w3, K, lane);
  for (int k = 32; k < K; k += 32) {
    __builtin_prefetch(arow + k + 32, 0, 1);
    bf16x16 an = load_a_f32(arow + k, K, lane);
    bf16x16 c0 = load_b_bf16(w0 + k, K, lane);
    bf16x16 c1 = load_b_bf16(w1 + k, K, lane);
    bf16x16 c2 = load_b_bf16(w2 + k, K, lane);
    bf16x16 c3 = load_b_bf16(w3 + k, K, lane);
    acc[0] = wmma_bf16(a, b0, acc[0]);
    acc[1] = wmma_bf16(a, b1, acc[1]);
    acc[2] = wmma_bf16(a, b2, acc[2]);
    acc[3] = wmma_bf16(a, b3, acc[3]);
    a = an; b0 = c0; b1 = c1; b2 = c2; b3 = c3;
  }
  acc[0] = wmma_bf16(a, b0, acc[0]);
  acc[1] = wmma_bf16(a, b1, acc[1]);
  acc[2] = wmma_bf16(a, b2, acc[2]);
  acc[3] = wmma_bf16(a, b3, acc[3]);
}

// ---------------------------------------------------------------------------
// C[M,N] = A1@W1^T (+ A2@W2^T) + b1 (+ b2); all operands pre-converted bf16.
// Block = 128 threads (4 waves); block tile 64x64; wave tile 16x64.
__global__ __launch_bounds__(128)
void gemm_dual_wmma(const __bf16* __restrict__ A1, const __bf16* __restrict__ W1, int K1,
                    const __bf16* __restrict__ A2, const __bf16* __restrict__ W2, int K2,
                    const float* __restrict__ bias1, const float* __restrict__ bias2,
                    float* __restrict__ C, int N) {
  const int lane = threadIdx.x & 31;
  const int wave = threadIdx.x >> 5;
  const int n0 = blockIdx.x * 64;
  const int m0 = blockIdx.y * 64 + wave * 16;

  f32x8 acc[4] = {};
  gemm_loop_bf16(A1, W1, K1, m0, n0, lane, acc);
  if (A2 != nullptr) gemm_loop_bf16(A2, W2, K2, m0, n0, lane, acc);

  // C layout: element e: lanes 0-15 -> row m0+e, lanes 16-31 -> row m0+8+e
  const int nl = lane & 15;
  const int ro = (lane >> 4) << 3;
#pragma unroll
  for (int j = 0; j < 4; ++j) {
    const int col = n0 + 16 * j + nl;
    float bv = 0.0f;
    if (bias1) bv += bias1[col];
    if (bias2) bv += bias2[col];
#pragma unroll
    for (int e = 0; e < 8; ++e) {
      C[(size_t)(m0 + ro + e) * N + col] = acc[j][e] + bv;
    }
  }
}

// ---------------------------------------------------------------------------
// Fused big GEMM: vpre = features[B*T,H] @ W_v^T ; then
// base_atten[row] += sum_col tanh(tanh(vpre+b_v) + h_t[b,col]) * W_av[col].
// One 64-row block == one batch's T rows, so b = blockIdx.y.
__global__ __launch_bounds__(128)
void gemm_v_attn_fused(const float* __restrict__ feats, const __bf16* __restrict__ Wv,
                       const float* __restrict__ bv, const float* __restrict__ ht,
                       const float* __restrict__ Wav, float* __restrict__ base_atten) {
  const int K = Hc;
  const int lane = threadIdx.x & 31;
  const int wave = threadIdx.x >> 5;
  const int n0 = blockIdx.x * 64;
  const int m0 = blockIdx.y * 64 + wave * 16;
  const int bidx = blockIdx.y;

  f32x8 acc[4] = {};
  gemm_loop_f32a(feats, Wv, K, m0, n0, lane, acc);

  const int nl = lane & 15;
  const int ro = (lane >> 4) << 3;
  float rowsum[8] = {0.f, 0.f, 0.f, 0.f, 0.f, 0.f, 0.f, 0.f};
#pragma unroll
  for (int j = 0; j < 4; ++j) {
    const int col = n0 + 16 * j + nl;
    const float bb   = bv[col];
    const float hcol = ht[(size_t)bidx * Hc + col];
    const float wav  = Wav[col];
#pragma unroll
    for (int e = 0; e < 8; ++e) {
      rowsum[e] += tanhf(tanhf(acc[j][e] + bb) + hcol) * wav;
    }
  }
  // Reduce the 16 lanes of each half-wave (16 cols of one row), then atomic.
#pragma unroll
  for (int e = 0; e < 8; ++e) {
    float s = rowsum[e];
    s += __shfl_xor(s, 1, 32);
    s += __shfl_xor(s, 2, 32);
    s += __shfl_xor(s, 4, 32);
    s += __shfl_xor(s, 8, 32);
    if (nl == 0) atomicAdd(&base_atten[m0 + ro + e], s);
  }
}

// ---------------------------------------------------------------------------
// f32 -> bf16 bulk conversion, 8 elements/thread, b128 in / b128 out.
__global__ void cvt_bf16(const float* __restrict__ s, __bf16* __restrict__ d, int n) {
  int i = (blockIdx.x * blockDim.x + threadIdx.x) * 8;
  if (i >= n) return;
  float4 a = *(const float4*)(s + i);
  float4 b = *(const float4*)(s + i + 4);
  bf16x8 o;
  o[0] = (__bf16)a.x; o[1] = (__bf16)a.y; o[2] = (__bf16)a.z; o[3] = (__bf16)a.w;
  o[4] = (__bf16)b.x; o[5] = (__bf16)b.y; o[6] = (__bf16)b.z; o[7] = (__bf16)b.w;
  *(bf16x8*)(d + i) = o;
}

__global__ void zero_f32(float* __restrict__ p, int n) {
  int i = blockIdx.x * blockDim.x + threadIdx.x;
  if (i < n) p[i] = 0.0f;
}

// LSTM pointwise; fuses the bf16 conversion of hy and sentinel for later GEMMs.
__global__ void lstm_pointwise(const float* __restrict__ gp, const float* __restrict__ spre,
                               const float* __restrict__ cx, float* __restrict__ hy,
                               __bf16* __restrict__ hyb, float* __restrict__ sent,
                               __bf16* __restrict__ sentb) {
  int idx = blockIdx.x * blockDim.x + threadIdx.x;  // over B*H
  int b = idx >> 10;
  int h = idx & (Hc - 1);
  const float* g = gp + (size_t)b * (4 * Hc);
  float i_ = sigm(g[h]);
  float f_ = sigm(g[Hc + h]);
  float g_ = tanhf(g[2 * Hc + h]);
  float o_ = sigm(g[3 * Hc + h]);
  float c  = f_ * cx[idx] + i_ * g_;
  float tc = tanhf(c);
  float hv = o_ * tc;
  float sv = sigm(spre[idx]) * tc;
  hy[idx] = hv;   hyb[idx]   = (__bf16)hv;
  sent[idx] = sv; sentb[idx] = (__bf16)sv;
}

// sentinel_atten[b] = sum_h tanh(ht[b,h]+st[b,h]) * W_as[h] + b_as
__global__ void sentinel_dot(const float* __restrict__ ht, const float* __restrict__ st,
                             const float* __restrict__ Was, const float* __restrict__ bas,
                             float* __restrict__ satt) {
  int b = blockIdx.x;
  int t = threadIdx.x;
  float s = 0.0f;
  for (int h = t; h < Hc; h += 256) {
    size_t idx = (size_t)b * Hc + h;
    s += tanhf(ht[idx] + st[idx]) * Was[h];
  }
  __shared__ float red[256];
  red[t] = s;
  __syncthreads();
  for (int w = 128; w > 0; w >>= 1) {
    if (t < w) red[t] += red[t + w];
    __syncthreads();
  }
  if (t == 0) satt[b] = red[0] + bas[0];
}

// softmax over 65 logits per batch: [base_atten[b,0..63]+b_av , satt[b]]
__global__ void softmax65(const float* __restrict__ base, const float* __restrict__ bav,
                          const float* __restrict__ satt, float* __restrict__ attn) {
  int b = blockIdx.x;
  int t = threadIdx.x;  // blockDim = 128, first 65 active
  __shared__ float l[65];
  __shared__ float mx, inv;
  if (t < 64) l[t] = base[b * Tc + t] + bav[0];
  else if (t == 64) l[64] = satt[b];
  __syncthreads();
  if (t == 0) {
    float m = -3.0e38f;
    for (int i = 0; i < 65; ++i) m = fmaxf(m, l[i]);
    float s = 0.0f;
    for (int i = 0; i < 65; ++i) s += expf(l[i] - m);
    mx = m;
    inv = 1.0f / s;
  }
  __syncthreads();
  if (t < 65) attn[b * 65 + t] = expf(l[t] - mx) * inv;
}

// cy_hat[b,h] = beta*sent + (1-beta)*(sum_t attn[b,t]*feat[b,t,h] + beta*sent)
__global__ void attend_out(const float* __restrict__ feats, const float* __restrict__ attn,
                           const float* __restrict__ sent, float* __restrict__ cyhat) {
  int b = blockIdx.x >> 2;
  int h = ((blockIdx.x & 3) << 8) + threadIdx.x;
  __shared__ float a[65];
  if (threadIdx.x < 65) a[threadIdx.x] = attn[b * 65 + threadIdx.x];
  __syncthreads();
  const float* fp = feats + (size_t)b * Tc * Hc + h;
  float s = 0.0f;
#pragma unroll 4
  for (int t = 0; t < Tc; ++t) s += a[t] * fp[(size_t)t * Hc];
  float beta = a[64];
  float sn = sent[(size_t)b * Hc + h];
  float visual = s + beta * sn;
  cyhat[(size_t)b * Hc + h] = beta * sn + (1.0f - beta) * visual;
}

// ---------------------------------------------------------------------------
extern "C" void kernel_launch(void* const* d_in, const int* in_sizes, int n_in,
                              void* d_out, int out_size, void* d_ws, size_t ws_size,
                              hipStream_t stream) {
  (void)in_sizes; (void)n_in; (void)out_size; (void)ws_size;
  const float* inputs   = (const float*)d_in[0];
  const float* hx       = (const float*)d_in[1];
  const float* cx       = (const float*)d_in[2];
  const float* features = (const float*)d_in[3];
  const float* W_ih = (const float*)d_in[4];
  const float* b_ih = (const float*)d_in[5];
  const float* W_hh = (const float*)d_in[6];
  const float* b_hh = (const float*)d_in[7];
  const float* W_si = (const float*)d_in[8];
  const float* b_si = (const float*)d_in[9];
  const float* W_sh = (const float*)d_in[10];
  const float* b_sh = (const float*)d_in[11];
  const float* W_st = (const float*)d_in[12];
  const float* b_st = (const float*)d_in[13];
  const float* W_ht = (const float*)d_in[14];
  const float* b_ht = (const float*)d_in[15];
  const float* W_v  = (const float*)d_in[16];
  const float* b_v  = (const float*)d_in[17];
  const float* W_as = (const float*)d_in[18];
  const float* b_as = (const float*)d_in[19];
  const float* W_av = (const float*)d_in[20];
  const float* b_av = (const float*)d_in[21];

  const size_t BH = (size_t)Bc * Hc;       // 1M
  // f32 scratch region
  float* ws   = (float*)d_ws;
  float* gp   = ws;            // [B,4H] (16 MB), reused after lstm_pointwise
  float* spre = ws + 4 * BH;   // [B,H]
  float* sent = ws + 5 * BH;   // [B,H]
  float* ht   = ws;            // reuse gp
  float* st   = ws + BH;
  float* base = ws + 2 * BH;                // [B,T]
  float* satt = base + (size_t)Bc * Tc;     // [B]
  float* attn = satt + Bc;                  // [B,65]
  // bf16 scratch region (starts after 6*BH floats = 24MB; 32B aligned)
  __bf16* bws    = (__bf16*)(ws + 6 * BH);
  __bf16* inp_b  = bws;                // BH
  __bf16* hx_b   = inp_b + BH;         // BH
  __bf16* hy_b   = hx_b + BH;          // BH
  __bf16* sent_b = hy_b + BH;          // BH
  __bf16* Wih_b  = sent_b + BH;        // 4*BH
  __bf16* Whh_b  = Wih_b + 4 * BH;     // 4*BH
  __bf16* Wsi_b  = Whh_b + 4 * BH;     // BH each below
  __bf16* Wsh_b  = Wsi_b + BH;
  __bf16* Wst_b  = Wsh_b + BH;
  __bf16* Wht_b  = Wst_b + BH;
  __bf16* Wv_b   = Wht_b + BH;

  float* hy    = (float*)d_out;
  float* cyhat = (float*)d_out + BH;

  const dim3 gblk(128);
  const int CT = 256;
  const int NB1 = (int)(BH / (8 * CT));  // blocks to convert 1M elements

  // One-time bf16 conversions (weights are L2-resident and reused ~1024x)
  cvt_bf16<<<NB1, CT, 0, stream>>>(inputs, inp_b, (int)BH);
  cvt_bf16<<<NB1, CT, 0, stream>>>(hx, hx_b, (int)BH);
  cvt_bf16<<<4 * NB1, CT, 0, stream>>>(W_ih, Wih_b, (int)(4 * BH));
  cvt_bf16<<<4 * NB1, CT, 0, stream>>>(W_hh, Whh_b, (int)(4 * BH));
  cvt_bf16<<<NB1, CT, 0, stream>>>(W_si, Wsi_b, (int)BH);
  cvt_bf16<<<NB1, CT, 0, stream>>>(W_sh, Wsh_b, (int)BH);
  cvt_bf16<<<NB1, CT, 0, stream>>>(W_st, Wst_b, (int)BH);
  cvt_bf16<<<NB1, CT, 0, stream>>>(W_ht, Wht_b, (int)BH);
  cvt_bf16<<<NB1, CT, 0, stream>>>(W_v, Wv_b, (int)BH);

  // gates = inputs@W_ih^T + hx@W_hh^T + b_ih + b_hh   (N=4096)
  gemm_dual_wmma<<<dim3(4 * Hc / 64, Bc / 64), gblk, 0, stream>>>(
      inp_b, Wih_b, Ec, hx_b, Whh_b, Hc, b_ih, b_hh, gp, 4 * Hc);
  // spre = inputs@W_si^T + hx@W_sh^T + b_si + b_sh
  gemm_dual_wmma<<<dim3(Hc / 64, Bc / 64), gblk, 0, stream>>>(
      inp_b, Wsi_b, Ec, hx_b, Wsh_b, Hc, b_si, b_sh, spre, Hc);
  // pointwise LSTM + sentinel (+ fused bf16 conversions)
  lstm_pointwise<<<(int)(BH / 256), 256, 0, stream>>>(gp, spre, cx, hy, hy_b, sent, sent_b);
  // h_t = hy@W_ht^T + b_ht ; s_t = sent@W_st^T + b_st
  gemm_dual_wmma<<<dim3(Hc / 64, Bc / 64), gblk, 0, stream>>>(
      hy_b, Wht_b, Hc, nullptr, nullptr, 0, b_ht, nullptr, ht, Hc);
  gemm_dual_wmma<<<dim3(Hc / 64, Bc / 64), gblk, 0, stream>>>(
      sent_b, Wst_b, Hc, nullptr, nullptr, 0, b_st, nullptr, st, Hc);
  // base_atten accumulation target must start at zero each call
  zero_f32<<<(Bc * Tc) / 256, 256, 0, stream>>>(base, Bc * Tc);
  // big fused GEMM + attention logit epilogue (M = B*T = 65536 rows)
  gemm_v_attn_fused<<<dim3(Hc / 64, (Bc * Tc) / 64), gblk, 0, stream>>>(
      features, Wv_b, b_v, ht, W_av, base);
  // sentinel attention logit
  sentinel_dot<<<Bc, 256, 0, stream>>>(ht, st, W_as, b_as, satt);
  // softmax over T+1
  softmax65<<<Bc, 128, 0, stream>>>(base, b_av, satt, attn);
  // weighted feature sum -> cy_hat
  attend_out<<<Bc * (Hc / 256), 256, 0, stream>>>(features, attn, sent, cyhat);
}
